// MultiHeadSelfAttention_15771119910962
// MI455X (gfx1250) — matrix operand order repaired and checked by reading, not compiled, not run on hardware
//
#include <hip/hip_runtime.h>
#include <hip/hip_bf16.h>
#include <math.h>

// ---------------------------------------------------------------------------
// MultiHeadSelfAttention for MI455X (gfx1250), bf16 WMMA + fp32 accumulate.
//   B=2, L=2048, H=512, NUM_HEADS=8, HEAD=64, ALIBI on first 4 heads.
// Kernel 1: QKV = x@W + bias  -> ws as bf16: Q[b,h,l,64] (pre-scaled 1/8),
//           K[b,h,l,64], Vt[b,h,64,l] (transposed for contiguous PV B-frags).
//           Epilogue is branchless (per-lane select of base/index/scale).
// Kernel 2: flash attention, 16 Q-rows per wave, 32-key chunks, online
//           softmax, LDS P-transpose, adj prefetch, non-temporal output.
// ---------------------------------------------------------------------------

typedef __attribute__((ext_vector_type(16))) __bf16 v16bf;
typedef __attribute__((ext_vector_type(8)))  __bf16 v8bf;
typedef __attribute__((ext_vector_type(8)))  float  v8f;

#define NHEADS 8
#define HEADD  64
#define LSEQ   2048
#define HID    512
#define B3H    1536
#define LOG2E  1.4426950408889634f

__device__ __forceinline__ v16bf join8(v8bf a, v8bf b) {
  return __builtin_shufflevector(a, b, 0,1,2,3,4,5,6,7,8,9,10,11,12,13,14,15);
}

// ---------------------------------------------------------------------------
// Kernel 1: QKV projection. Grid (24, 256); 128 threads = 4 waves, each wave
// owns one 16-col tile of the 1536 output columns; blockIdx.y = 16-row tile
// over flattened [B*L, 512] rows.
// ---------------------------------------------------------------------------
__global__ __launch_bounds__(128)
void qkv_proj_kernel(const float* __restrict__ x, const float* __restrict__ w,
                     const float* __restrict__ in_bias,
                     void* qws, void* kws, void* vtws)
{
  __bf16* Qw  = (__bf16*)qws;
  __bf16* Kw  = (__bf16*)kws;
  __bf16* Vtw = (__bf16*)vtws;

  const int lane  = threadIdx.x & 31;
  const int wv    = threadIdx.x >> 5;
  const int m0    = blockIdx.y * 16;              // flattened row tile (B*L)
  const int ncol0 = (blockIdx.x * 4 + wv) * 16;   // output col tile

  const int arow  = lane & 15;                    // A: lane = M row
  const int kb    = (lane >> 4) * 8;              // A: K sub-base per half
  const int koff  = (lane >> 4) * 16;             // B: K offset per half
  const int bcol  = ncol0 + (lane & 15);          // B: lane = N column

  v8f acc = {};
  const float* xrow = x + (size_t)(m0 + arow) * HID;
  for (int kc = 0; kc < HID; kc += 32) {
    // A fragment: x 16x32 tile, per-lane two contiguous 8-float chunks.
    const float* pa = xrow + kc + kb;
    v16bf A;
    #pragma unroll
    for (int j = 0; j < 8; ++j) {
      A[j]     = (__bf16)pa[j];        // K = kb + j
      A[8 + j] = (__bf16)pa[16 + j];   // K = kb + 16 + j
    }
    // B fragment: W 32x16 tile (column per lane; stride 1536 per K step).
    v16bf Bf;
    #pragma unroll
    for (int j = 0; j < 16; ++j)
      Bf[j] = (__bf16)w[(size_t)(kc + kb + j + (j >= 8 ? 8 : 0)) * B3H + bcol];
    acc = __builtin_amdgcn_wmma_f32_16x16x32_bf16(false, A, false, Bf,
                                                  (short)0, acc, false, false);
  }

  // ---- branchless epilogue: one store path, per-lane selected dest ----
  const float bias = in_bias[bcol];
  const int hh    = bcol / 192;                   // head
  const int r3    = bcol - hh * 192;
  const int which = r3 >> 6;                      // 0=Q 1=K 2=V
  const int dd    = r3 & 63;
  const float vscale = (which == 0) ? 0.125f : 1.0f;   // fold 1/sqrt(64) into Q
  __bf16* basep = (which == 0) ? Qw : ((which == 1) ? Kw : Vtw);
  #pragma unroll
  for (int r = 0; r < 8; ++r) {
    const int g  = m0 + (lane >> 4) * 8 + r;      // flattened row (C layout)
    const int bb = g >> 11;
    const int ll = g & 2047;
    const size_t bh = (size_t)bb * NHEADS + hh;
    // Q/K: [bh][l][64] ; Vt: [bh][64][L]
    const size_t idx = (which == 2) ? (bh * HEADD + dd) * LSEQ + ll
                                    : (bh * LSEQ + ll) * HEADD + dd;
    basep[idx] = (__bf16)((acc[r] + bias) * vscale);
  }
}

// ---------------------------------------------------------------------------
// Kernel 2: flash attention. Grid (32, 8, 2); 128 threads = 4 waves; each
// wave owns a 16-row Q tile and streams all 2048 keys in 32-key chunks.
// ---------------------------------------------------------------------------
__global__ __launch_bounds__(128)
void flash_attn_kernel(const void* qws, const void* kws, const void* vtws,
                       const float* __restrict__ adj,
                       const float* __restrict__ gamma,
                       const float* __restrict__ out_bias,
                       float* __restrict__ out)
{
  __shared__ __align__(16) __bf16 pbuf[4][16][40];  // P transpose staging/wave

  const int lane = threadIdx.x & 31;
  const int wv   = threadIdx.x >> 5;
  const int h    = blockIdx.y;
  const int bb   = blockIdx.z;
  const int i0   = (blockIdx.x * 4 + wv) * 16;

  const size_t qk = ((size_t)bb * NHEADS + h) * (size_t)LSEQ * HEADD;
  const __bf16* Q  = (const __bf16*)qws  + qk;
  const __bf16* K  = (const __bf16*)kws  + qk;
  const __bf16* Vt = (const __bf16*)vtws + qk;     // [64][2048]
  const float*  adjb = adj + (size_t)bb * LSEQ * LSEQ;

  const int arow  = lane & 15;
  const int kb    = (lane >> 4) * 8;
  const int koff  = (lane >> 4) * 16;
  const int bcol  = lane & 15;
  const int rbase = (lane >> 4) * 8;

  const float slope = (h < 4) ? exp2f(-2.0f * (float)(h + 1)) : 0.0f;
  const float gam   = gamma[h];

  // Q A-fragments for K(head-dim)=0..31 and 32..63 (scale already folded in).
  v16bf qA[2];
  #pragma unroll
  for (int dh = 0; dh < 2; ++dh) {
    const __bf16* p = Q + (size_t)(i0 + arow) * HEADD + dh * 32 + kb;
    qA[dh] = join8(*(const v8bf*)p, *(const v8bf*)(p + 16));
  }

  float m[8], lsum[8];
  #pragma unroll
  for (int r = 0; r < 8; ++r) { m[r] = -3.0e38f; lsum[r] = 0.0f; }
  v8f O[4] = {};   // 16x64 output accumulator (4 col groups)

  for (int j0 = 0; j0 < LSEQ; j0 += 32) {
    // ---- S = Q*K^T for 16x32 score tile (two 16-col groups) ----
    v8f s[2];
    #pragma unroll
    for (int jg = 0; jg < 2; ++jg) {
      v8f t = {};
      #pragma unroll
      for (int dh = 0; dh < 2; ++dh) {
        const __bf16* p = K + (size_t)(j0 + jg*16 + bcol) * HEADD + dh*32 + kb;
        v16bf kB = join8(*(const v8bf*)p, *(const v8bf*)(p + 16));
        t = __builtin_amdgcn_wmma_f32_16x16x32_bf16(false, qA[dh], false, kB,
                                                    (short)0, t, false, false);
      }
      // alibi + gamma*adj bias in C-tile layout; prefetch next chunk's adj
      const int j = j0 + jg*16 + bcol;
      #pragma unroll
      for (int r = 0; r < 8; ++r) {
        const int i = i0 + rbase + r;
        const float* ap = adjb + (size_t)i * LSEQ + j;
        t[r] += gam * ap[0] - slope * fabsf((float)(j - i));
        if (j0 + 32 < LSEQ)
          __builtin_prefetch(ap + 32, 0, 1);   // global_prefetch_b8, next chunk
      }
      s[jg] = t;
    }

    // ---- online softmax (rows striped: reduce across 16-lane halves) ----
    float scl[8];
    #pragma unroll
    for (int r = 0; r < 8; ++r) {
      float mm = fmaxf(s[0][r], s[1][r]);
      mm = fmaxf(mm, __shfl_xor(mm, 1));
      mm = fmaxf(mm, __shfl_xor(mm, 2));
      mm = fmaxf(mm, __shfl_xor(mm, 4));
      mm = fmaxf(mm, __shfl_xor(mm, 8));
      const float mn = fmaxf(m[r], mm);
      scl[r] = exp2f((m[r] - mn) * LOG2E);
      m[r] = mn;
    }
    #pragma unroll
    for (int jg = 0; jg < 2; ++jg) {
      #pragma unroll
      for (int r = 0; r < 8; ++r) {
        const float p = exp2f((s[jg][r] - m[r]) * LOG2E);
        s[jg][r] = p;
        pbuf[wv][rbase + r][jg*16 + bcol] = (__bf16)p;  // C-layout -> LDS
      }
    }
    #pragma unroll
    for (int r = 0; r < 8; ++r) {
      float rs = s[0][r] + s[1][r];
      rs += __shfl_xor(rs, 1);
      rs += __shfl_xor(rs, 2);
      rs += __shfl_xor(rs, 4);
      rs += __shfl_xor(rs, 8);
      lsum[r] = lsum[r] * scl[r] + rs;
    }
    #pragma unroll
    for (int dg = 0; dg < 4; ++dg)
      #pragma unroll
      for (int r = 0; r < 8; ++r)
        O[dg][r] *= scl[r];

    // LDS stores complete before layout-transposed reads (same-wave fence).
    asm volatile("s_wait_dscnt 0" ::: "memory");

    // P as A-fragment (16x32, K = key chunk), read back from LDS.
    const __bf16* pp = &pbuf[wv][arow][kb];
    v16bf pA = join8(*(const v8bf*)pp, *(const v8bf*)(pp + 16));

    // ---- O += P * V : 4 WMMAs against contiguous Vt B-fragments ----
    #pragma unroll
    for (int dg = 0; dg < 4; ++dg) {
      const __bf16* pv = Vt + (size_t)(dg*16 + bcol) * LSEQ + j0 + kb;
      v16bf vB = join8(*(const v8bf*)pv, *(const v8bf*)(pv + 16));
      O[dg] = __builtin_amdgcn_wmma_f32_16x16x32_bf16(false, pA, false, vB,
                                                      (short)0, O[dg], false, false);
    }
  }

  // ---- epilogue: normalize, add out_bias, non-temporal fp32 store ----
  float inv[8];
  #pragma unroll
  for (int r = 0; r < 8; ++r) inv[r] = 1.0f / lsum[r];
  #pragma unroll
  for (int dg = 0; dg < 4; ++dg) {
    const int col = h * HEADD + dg*16 + bcol;
    const float ob = out_bias[col];
    #pragma unroll
    for (int r = 0; r < 8; ++r) {
      const int i = i0 + rbase + r;
      __builtin_nontemporal_store(O[dg][r] * inv[r] + ob,
                                  out + ((size_t)bb * LSEQ + i) * HID + col);
    }
  }
}

// ---------------------------------------------------------------------------
extern "C" void kernel_launch(void* const* d_in, const int* in_sizes, int n_in,
                              void* d_out, int out_size, void* d_ws, size_t ws_size,
                              hipStream_t stream) {
  (void)in_sizes; (void)n_in; (void)out_size; (void)ws_size;
  const float* x        = (const float*)d_in[0];
  const float* adj      = (const float*)d_in[1];
  const float* weights  = (const float*)d_in[2];
  const float* in_bias  = (const float*)d_in[3];
  const float* out_bias = (const float*)d_in[4];
  const float* gamma    = (const float*)d_in[5];
  float* out = (float*)d_out;

  // ws: Q (4MB) | K (4MB) | Vt (4MB), bf16
  const size_t tbytes = (size_t)2 * NHEADS * LSEQ * HEADD * 2;
  char* ws = (char*)d_ws;
  void* Qw  = ws;
  void* Kw  = ws + tbytes;
  void* Vtw = ws + 2 * tbytes;

  dim3 g1(B3H / 64, (2 * LSEQ) / 16, 1);          // (24, 256)
  qkv_proj_kernel<<<g1, 128, 0, stream>>>(x, weights, in_bias, Qw, Kw, Vtw);

  dim3 g2(LSEQ / 64, NHEADS, 2);                   // (32, 8, 2)
  flash_attn_kernel<<<g2, 128, 0, stream>>>(Qw, Kw, Vtw, adj, gamma, out_bias, out);
}
